// STFTLEARN_85289460564285
// MI455X (gfx1250) — matrix operand order repaired and checked by reading, not compiled, not run on hardware
//
#include <hip/hip_runtime.h>
#include <hip/hip_bf16.h>

typedef __attribute__((ext_vector_type(16))) _Float16 v16h;
typedef __attribute__((ext_vector_type(8)))  _Float16 v8h;
typedef __attribute__((ext_vector_type(8)))  float    v8f;

#define FILTER_LEN 1024
#define HOP        512
#define CUTOFF     513
#define NSAMP      491520
#define NFRAMES    961
#define NBATCH     32
#define MROWS      1026                  // 513 real + 513 imag (interleaved after packing)
#define MPAD       1088                  // 17 * 64
#define ROWBLOCKS  (MPAD / 16)           // 68
#define KBLOCKS    (FILTER_LEN / 32)     // 32
#define APACK_HALFS (ROWBLOCKS * KBLOCKS * 32 * 16)
#define BSTRIDE    40                    // LDS row stride in halfs (bank-conflict padding)
#define TFRM       64                    // frames per workgroup tile

// ---------------------------------------------------------------------------
// Pack W = forward_basis * window into WMMA A-matrix (16x32 f16) lane layout,
// split into hi/lo f16 for split-precision (Markidis) accumulation.
// Packed row p: even p -> real row p/2, odd p -> imag row 513 + p/2, so that
// (re, im) of one frequency land in adjacent accumulator VGPRs of one lane.
// Layout index: ((rb*KBLOCKS + kb)*32 + lane)*16 + h, with a contiguous-k
// 16-bit A 16x32 mapping: lane = g*16 + (m&15);
// k_local = g*16 + h.
// ---------------------------------------------------------------------------
__global__ void stft_pack_basis(const float* __restrict__ fb,
                                const float* __restrict__ win,
                                _Float16* __restrict__ Ah,
                                _Float16* __restrict__ Al) {
    int idx = blockIdx.x * blockDim.x + threadIdx.x;
    if (idx >= APACK_HALFS) return;
    int h  = idx & 15;
    int l  = (idx >> 4) & 31;
    int kb = (idx >> 9) & 31;
    int rb = idx >> 14;

    int g  = l >> 4;
    int lm = l & 15;
    int p  = rb * 16 + lm;
    int hv = h >> 1, hp = h & 1;
    int klocal = g * 16 + h;
    int k = kb * 32 + klocal;

    float w = 0.0f;
    if (p < MROWS) {
        int f = (p & 1) ? (CUTOFF + (p >> 1)) : (p >> 1);
        w = fb[f * FILTER_LEN + k] * win[k];
    }
    _Float16 hi = (_Float16)w;
    _Float16 lo = (_Float16)(w - (float)hi);
    Ah[idx] = hi;
    Al[idx] = lo;
}

// ---------------------------------------------------------------------------
// WMMA STFT GEMM. Workgroup tile: 64 packed rows x 64 frames, 8 waves.
// Each wave: one A fragment (16 rows) reused across TWO 16-frame N tiles
// -> 6 WMMAs per K-step per wave on 4 independent accumulators
// (hh0, hh1, corr0, corr1) so adjacent WMMAs never stall on D->C chains.
// Split-f16: acc = Ah*Bh + (Ah*Bl + Al*Bh), f32 accumulate.
// ---------------------------------------------------------------------------
__global__ void __launch_bounds__(256)
stft_wmma(const float* __restrict__ x,
          const _Float16* __restrict__ Ah,
          const _Float16* __restrict__ Al,
          float* __restrict__ out) {
    __shared__ _Float16 sBh[TFRM * BSTRIDE];
    __shared__ _Float16 sBl[TFRM * BSTRIDE];

    const int b  = blockIdx.z;
    const int m0 = blockIdx.y * 64;      // packed-row base
    const int t0 = blockIdx.x * TFRM;    // frame base

    const int tid  = threadIdx.x;
    const int lane = tid & 31;
    const int wave = tid >> 5;           // 0..7
    const int mi   = wave >> 1;          // 0..3 (row sub-tile)
    const int ni   = wave & 1;           // 0..1 (32-frame half)

    const int rbGlobal = (m0 >> 4) + mi;

    // B staging: thread -> (frame, 8 consecutive k); 64 frames x 32 k per step
    const int  bn      = tid >> 2;               // 0..63
    const int  bk      = (tid & 3) * 8;          // 0,8,16,24
    const long sigBase = (long)b * NSAMP;
    const int  gBase   = (t0 + bn) * HOP + bk;   // padded-signal index base

    v8f acc0h = {}, acc0c = {}, acc1h = {}, acc1c = {};

    for (int kb = 0; kb < KBLOCKS; ++kb) {
        // Load 8 input samples with reflect padding (PAD = 512)
        float v[8];
#pragma unroll
        for (int c = 0; c < 8; ++c) {
            int s = gBase + kb * 32 + c - 512;
            s = (s < 0) ? -s : s;
            s = (s >= NSAMP) ? (2 * NSAMP - 2 - s) : s;
            v[c] = x[sigBase + s];
        }

        __syncthreads();                 // prior iteration's LDS reads done
#pragma unroll
        for (int c = 0; c < 8; ++c) {
            _Float16 hi = (_Float16)v[c];
            _Float16 lo = (_Float16)(v[c] - (float)hi);
            sBh[bn * BSTRIDE + bk + c] = hi;
            sBl[bn * BSTRIDE + bk + c] = lo;
        }
        __syncthreads();                 // stores visible to all waves

        // A fragments: pre-swizzled, 16 contiguous halfs (32B) per lane
        const int aoff = ((rbGlobal * KBLOCKS + kb) * 32 + lane) * 16;
        v16h afh = *reinterpret_cast<const v16h*>(Ah + aoff);
        v16h afl = *reinterpret_cast<const v16h*>(Al + aoff);
        if (kb + 1 < KBLOCKS) {
            __builtin_prefetch(Ah + aoff + 512, 0, 1);   // next K-block
            __builtin_prefetch(Al + aoff + 512, 0, 1);
        }

        // B fragments for two adjacent 16-frame tiles.
        // Col n = lane&15; lanes 0-15 hold K 0..15, lanes 16-31 hold K 16..31.
        const int nlA  = ni * 32 + (lane & 15);
        const int gsel = lane >> 4;
        const int offA = nlA * BSTRIDE + gsel * 16;
        const int offB = offA + 16 * BSTRIDE;
        v8h h0a = *reinterpret_cast<const v8h*>(sBh + offA);
        v8h h1a = *reinterpret_cast<const v8h*>(sBh + offA + 8);
        v8h l0a = *reinterpret_cast<const v8h*>(sBl + offA);
        v8h l1a = *reinterpret_cast<const v8h*>(sBl + offA + 8);
        v8h h0b = *reinterpret_cast<const v8h*>(sBh + offB);
        v8h h1b = *reinterpret_cast<const v8h*>(sBh + offB + 8);
        v8h l0b = *reinterpret_cast<const v8h*>(sBl + offB);
        v8h l1b = *reinterpret_cast<const v8h*>(sBl + offB + 8);
        v16h b0h, b0l, b1h, b1l;
#pragma unroll
        for (int i = 0; i < 8; ++i) {
            b0h[i] = h0a[i]; b0h[8 + i] = h1a[i];
            b0l[i] = l0a[i]; b0l[8 + i] = l1a[i];
            b1h[i] = h0b[i]; b1h[8 + i] = h1b[i];
            b1l[i] = l0b[i]; b1l[8 + i] = l1b[i];
        }

        // 6 WMMAs, 4 independent accumulator chains (no back-to-back D->C RAW)
        acc0h = __builtin_amdgcn_wmma_f32_16x16x32_f16(false, afh, false, b0h,
                                                       (short)0, acc0h, false, false);
        acc1h = __builtin_amdgcn_wmma_f32_16x16x32_f16(false, afh, false, b1h,
                                                       (short)0, acc1h, false, false);
        acc0c = __builtin_amdgcn_wmma_f32_16x16x32_f16(false, afh, false, b0l,
                                                       (short)0, acc0c, false, false);
        acc1c = __builtin_amdgcn_wmma_f32_16x16x32_f16(false, afh, false, b1l,
                                                       (short)0, acc1c, false, false);
        acc0c = __builtin_amdgcn_wmma_f32_16x16x32_f16(false, afl, false, b0h,
                                                       (short)0, acc0c, false, false);
        acc1c = __builtin_amdgcn_wmma_f32_16x16x32_f16(false, afl, false, b1h,
                                                       (short)0, acc1c, false, false);
    }

    // Epilogue: D layout -> lane (g, n) holds rows g*8+v, col n.
    // Packed rows interleave (real, imag): (acc[2i], acc[2i+1]) is one freq.
    const int nl = lane & 15;
    const int g  = lane >> 4;
    const long magBase = (long)b * CUTOFF * NFRAMES;
    const long phBase  = (long)NBATCH * CUTOFF * NFRAMES + magBase;

#pragma unroll
    for (int tile = 0; tile < 2; ++tile) {
        const int t = t0 + ni * 32 + tile * 16 + nl;
        if (t < NFRAMES) {
            const v8f& ah_ = tile ? acc1h : acc0h;
            const v8f& ac_ = tile ? acc1c : acc0c;
#pragma unroll
            for (int i = 0; i < 4; ++i) {
                int p  = m0 + mi * 16 + g * 8 + 2 * i;   // even packed row
                int fr = p >> 1;
                if (fr < CUTOFF) {
                    float re  = ah_[2 * i]     + ac_[2 * i];
                    float im  = ah_[2 * i + 1] + ac_[2 * i + 1];
                    float mag = sqrtf(re * re + im * im);
                    float ph  = atan2f(im, re);
                    out[magBase + (long)fr * NFRAMES + t] = mag;
                    out[phBase  + (long)fr * NFRAMES + t] = ph;
                }
            }
        }
    }
}

// ---------------------------------------------------------------------------
extern "C" void kernel_launch(void* const* d_in, const int* in_sizes, int n_in,
                              void* d_out, int out_size, void* d_ws, size_t ws_size,
                              hipStream_t stream) {
    const float* x   = (const float*)d_in[0];   // (32, 491520)
    const float* fb  = (const float*)d_in[1];   // (1026, 1024)
    const float* win = (const float*)d_in[2];   // (1024,)
    float* out = (float*)d_out;                 // mag (32,513,961) ++ phase

    _Float16* Ah = (_Float16*)d_ws;
    _Float16* Al = Ah + APACK_HALFS;            // ~4.5 MB total workspace

    stft_pack_basis<<<(APACK_HALFS + 255) / 256, 256, 0, stream>>>(fb, win, Ah, Al);

    dim3 grid((NFRAMES + TFRM - 1) / TFRM, MPAD / 64, NBATCH);   // 16 x 17 x 32
    stft_wmma<<<grid, 256, 0, stream>>>(x, Ah, Al, out);
}
